// TileGymLMHeadJSD_4183298146641
// MI455X (gfx1250) — matrix operand order repaired and checked by reading, not compiled, not run on hardware
//
#include <hip/hip_runtime.h>
#include <hip/hip_bf16.h>
#include <cstdint>
#include <cstddef>

// ---------------------------------------------------------------------------
// Fused LM-head JSD (Liger semantics) for MI455X / gfx1250, wave32 + WMMA.
//   BT=2048, H=1024, V=32000, beta=0.5, T=1.0, ignore_index=-100
// Pipeline:
//   1) gemm_jsd<1>: bf16 WMMA logits tiles (student+teacher), online per-row
//      (max, sumexp) partials per 64-wide vocab block  -> ws.partials
//   2) lse_reduce:  merge 500 partials per token -> logsumexp per token
//   3) gemm_jsd<2>: recompute logits tiles, accumulate JSD contribution per
//      workgroup -> ws.jsdPart (deterministic, no float atomics)
//   4) finalize:    sum partials, count non-ignored labels, write loss
//
// GEMM structure: 64x64 workgroup tile, 8 wave32s each owning 32 rows x 16
// cols (two A frags reuse one B frag -> 3 ds_load_b128 per wmma), K-step 64,
// double-buffered LDS, register-staged (pre-packed bf16) global loads with
// strength-reduced pointers for overlap.
// ---------------------------------------------------------------------------

typedef __attribute__((ext_vector_type(16))) __bf16 v16bf;
typedef __attribute__((ext_vector_type(8)))  float  v8f;

namespace {
constexpr int kBT = 2048;
constexpr int kH  = 1024;
constexpr int kV  = 32000;
constexpr int MT  = 64;    // tokens per workgroup
constexpr int NT  = 64;    // vocab cols per workgroup
constexpr int KS  = 64;    // K step (two 16x16x32 WMMAs)
constexpr int LDK = 72;    // padded LDS K stride (bf16 elems) -> 144B rows
constexpr int NSTAGE = kH / KS;  // 16 K stages
constexpr int kMB = kBT / MT;    // 32 token blocks
constexpr int kVB = kV / NT;     // 500 vocab blocks
constexpr float kBeta = 0.5f;
}

union FragBF {
  uint4 q[2];
  v16bf v;
};

static __device__ __forceinline__ unsigned pack_bf16(float a, float b) {
  unsigned ua = __float_as_uint(a), ub = __float_as_uint(b);
  ua = (ua + 0x7FFFu + ((ua >> 16) & 1u)) >> 16;  // round-to-nearest-even
  ub = (ub + 0x7FFFu + ((ub >> 16) & 1u)) >> 16;
  return ua | (ub << 16);
}

static __device__ __forceinline__ uint2 pack4(float4 f) {
  return make_uint2(pack_bf16(f.x, f.y), pack_bf16(f.z, f.w));
}

// Butterfly reductions. Masks 1,2,4,8 keep the two 16-lane halves of a
// wave32 independent (rows 0-7 vs rows 8-15 of a 16x16 C tile).
static __device__ __forceinline__ float rowmax16(float v) {
#pragma unroll
  for (int m = 1; m < 16; m <<= 1) v = fmaxf(v, __shfl_xor(v, m, 32));
  return v;
}
static __device__ __forceinline__ float rowsum16(float v) {
#pragma unroll
  for (int m = 1; m < 16; m <<= 1) v += __shfl_xor(v, m, 32);
  return v;
}
static __device__ __forceinline__ float sum32(float v) {
#pragma unroll
  for (int m = 1; m < 32; m <<= 1) v += __shfl_xor(v, m, 32);
  return v;
}

// A frag (16x32 bf16, ISA 7.12.2): lanes 0-15 hold K {0..7,16..23}, lanes
// 16-31 hold K {8..15,24..31} for their row.
static __device__ __forceinline__ FragBF load_fragA(const __bf16* tile,
                                                    int row, int kk, int kh) {
  FragBF f;
  const __bf16* p = tile + row * LDK + kk + kh * 8;
  f.q[0] = *(const uint4*)(p);
  f.q[1] = *(const uint4*)(p + 16);
  return f;
}
// B frag (32x16 bf16): lanes 0-15 hold K {0..7,16..23}, lanes 16-31 hold K
// {8..15,24..31} for their column; LDS W tile is [n][k] so K is contiguous per column.
static __device__ __forceinline__ FragBF load_fragB(const __bf16* tile,
                                                    int col, int kk, int kh) {
  FragBF f;
  const __bf16* p = tile + col * LDK + kk + kh * 8;
  f.q[0] = *(const uint4*)(p);
  f.q[1] = *(const uint4*)(p + 16);
  return f;
}

template <int PASS>
__global__ __launch_bounds__(256)
void gemm_jsd(const float* __restrict__ Xs, const float* __restrict__ Xt,
              const float* __restrict__ Ws, const float* __restrict__ Wt,
              const int* __restrict__ label,
              float4* __restrict__ partials,       // [kVB][kBT]  (pass 1 out)
              const float2* __restrict__ lse,      // [kBT]       (pass 2 in)
              float* __restrict__ jsdPart) {       // [kVB*kMB]   (pass 2 out)
  __shared__ __align__(16) __bf16 sAs[2][MT * LDK];
  __shared__ __align__(16) __bf16 sAt[2][MT * LDK];
  __shared__ __align__(16) __bf16 sBs[2][NT * LDK];
  __shared__ __align__(16) __bf16 sBt[2][NT * LDK];
  __shared__ float2 redS[MT][4];
  __shared__ float2 redT[MT][4];
  __shared__ float2 lseSh[MT];
  __shared__ float  maskSh[MT];
  __shared__ float  wsum[8];

  const int tid  = threadIdx.x;
  const int tok0 = blockIdx.x * MT;
  const int v0   = blockIdx.y * NT;

  if (PASS == 2 && tid < MT) {
    lseSh[tid]  = lse[tok0 + tid];
    maskSh[tid] = (label[tok0 + tid] != -100) ? 1.0f : 0.0f;
  }

  const int lane  = tid & 31;
  const int wv    = tid >> 5;      // wave 0..7
  const int kh    = lane >> 4;     // lane half
  const int idx   = lane & 15;
  const int mpair = wv >> 2;       // 0..1 : which 32-row half
  const int ncol  = wv & 3;        // 0..3 : which 16-col stripe

  // Per-wave output: rows mpair*32 + {0,16} + idx-row, cols ncol*16 + idx.
  v8f cS[2] = {{}, {}};
  v8f cT[2] = {{}, {}};

  // Per-thread staging slots: slot i covers row=srow+i*16, col=scol..scol+3
  // of the 64x64 f32 tile. Row stride 16*kH floats = 64KB, a compile-time
  // offset that folds into the global_load_b128 instruction offset; the base
  // pointers advance by KS floats per stage (strength reduction).
  const int srow = tid >> 4;
  const int scol = (tid & 15) * 4;
  const float* pAs = Xs + (size_t)(tok0 + srow) * kH + scol;
  const float* pAt = Xt + (size_t)(tok0 + srow) * kH + scol;
  const float* pBs = Ws + (size_t)(v0 + srow) * kH + scol;
  const float* pBt = Wt + (size_t)(v0 + srow) * kH + scol;

  uint2 rA_s[4], rA_t[4], rB_s[4], rB_t[4];  // staged, already packed bf16x4

  auto load_stage = [&]() {
#pragma unroll
    for (int i = 0; i < 4; ++i) {
      rA_s[i] = pack4(*(const float4*)&pAs[i * 16 * kH]);
      rA_t[i] = pack4(*(const float4*)&pAt[i * 16 * kH]);
      rB_s[i] = pack4(*(const float4*)&pBs[i * 16 * kH]);
      rB_t[i] = pack4(*(const float4*)&pBt[i * 16 * kH]);
    }
    pAs += KS; pAt += KS; pBs += KS; pBt += KS;
  };
  auto store_stage = [&](int b) {
#pragma unroll
    for (int i = 0; i < 4; ++i) {
      const int off = (srow + i * 16) * LDK + scol;
      *(uint2*)&sAs[b][off] = rA_s[i];
      *(uint2*)&sAt[b][off] = rA_t[i];
      *(uint2*)&sBs[b][off] = rB_s[i];
      *(uint2*)&sBt[b][off] = rB_t[i];
    }
  };

  // -------- software-pipelined main loop --------
  load_stage();
  store_stage(0);
  for (int s = 0; s < NSTAGE; ++s) {
    __syncthreads();  // buf[s&1] ready; all reads of buf[(s+1)&1] retired
    const int cur = s & 1;
    if (s + 1 < NSTAGE) load_stage();  // globals in flight behind WMMA

    const int arow = mpair * 32 + idx;
    const int bcol = ncol * 16 + idx;
#pragma unroll
    for (int kk = 0; kk < KS; kk += 32) {
      {  // student
        FragBF a0 = load_fragA(sAs[cur], arow, kk, kh);
        FragBF a1 = load_fragA(sAs[cur], arow + 16, kk, kh);
        FragBF b  = load_fragB(sBs[cur], bcol, kk, kh);
        cS[0] = __builtin_amdgcn_wmma_f32_16x16x32_bf16(
            false, a0.v, false, b.v, (short)0, cS[0], false, false);
        cS[1] = __builtin_amdgcn_wmma_f32_16x16x32_bf16(
            false, a1.v, false, b.v, (short)0, cS[1], false, false);
      }
      {  // teacher
        FragBF a0 = load_fragA(sAt[cur], arow, kk, kh);
        FragBF a1 = load_fragA(sAt[cur], arow + 16, kk, kh);
        FragBF b  = load_fragB(sBt[cur], bcol, kk, kh);
        cT[0] = __builtin_amdgcn_wmma_f32_16x16x32_bf16(
            false, a0.v, false, b.v, (short)0, cT[0], false, false);
        cT[1] = __builtin_amdgcn_wmma_f32_16x16x32_bf16(
            false, a1.v, false, b.v, (short)0, cT[1], false, false);
      }
    }
    if (s + 1 < NSTAGE) store_stage(1 - cur);
  }

  // C/D layout: lane holds col N=idx; VGPR r holds row M = r + 8*kh (7.12.2).
  if (PASS == 1) {
    // Per-row online softmax partials over this 64-wide vocab stripe.
#pragma unroll
    for (int t = 0; t < 2; ++t) {
#pragma unroll
      for (int r = 0; r < 8; ++r) {
        const int mloc = mpair * 32 + t * 16 + r + 8 * kh;
        const float vS = cS[t][r];
        const float mS = rowmax16(vS);
        const float eS = rowsum16(__expf(vS - mS));
        const float vT = cT[t][r];
        const float mT = rowmax16(vT);
        const float eT = rowsum16(__expf(vT - mT));
        if (idx == 0) {
          redS[mloc][ncol] = make_float2(mS, eS);
          redT[mloc][ncol] = make_float2(mT, eT);
        }
      }
    }
    __syncthreads();
    if (tid < MT) {
      float sM = -__builtin_inff(), tM = -__builtin_inff();
#pragma unroll
      for (int j = 0; j < 4; ++j) {
        sM = fmaxf(sM, redS[tid][j].x);
        tM = fmaxf(tM, redT[tid][j].x);
      }
      float sS = 0.f, tS = 0.f;
#pragma unroll
      for (int j = 0; j < 4; ++j) {
        sS += redS[tid][j].y * __expf(redS[tid][j].x - sM);
        tS += redT[tid][j].y * __expf(redT[tid][j].x - tM);
      }
      partials[(size_t)blockIdx.y * kBT + tok0 + tid] =
          make_float4(sM, sS, tM, tS);
    }
  } else {
    // JSD contribution of this tile (beta = 0.5).
    float csum = 0.f;
#pragma unroll
    for (int t = 0; t < 2; ++t) {
#pragma unroll
      for (int r = 0; r < 8; ++r) {
        const int mloc = mpair * 32 + t * 16 + r + 8 * kh;
        const float2 L = lseSh[mloc];
        const float msk = maskSh[mloc];
        const float lq = cS[t][r] - L.x;  // student log-prob
        const float lp = cT[t][r] - L.y;  // teacher log-prob
        const float q = __expf(lq);
        const float p = __expf(lp);
        const float lm = __logf(kBeta * p + (1.f - kBeta) * q);
        csum += msk * (kBeta * p * (lp - lm) + (1.f - kBeta) * q * (lq - lm));
      }
    }
    csum = sum32(csum);
    if (lane == 0) wsum[wv] = csum;
    __syncthreads();
    if (tid == 0) {
      float t = 0.f;
#pragma unroll
      for (int i = 0; i < 8; ++i) t += wsum[i];
      jsdPart[(size_t)blockIdx.y * kMB + blockIdx.x] = t;
    }
  }
}

__global__ __launch_bounds__(256)
void lse_reduce(const float4* __restrict__ partials, float2* __restrict__ lse) {
  const int t = blockIdx.x * blockDim.x + threadIdx.x;
  if (t >= kBT) return;
  float sM = -__builtin_inff(), sS = 0.f;
  float tM = -__builtin_inff(), tS = 0.f;
  for (int vb = 0; vb < kVB; ++vb) {
    const float4 p = partials[(size_t)vb * kBT + t];
    float nM = fmaxf(sM, p.x);
    sS = sS * __expf(sM - nM) + p.y * __expf(p.x - nM);
    sM = nM;
    nM = fmaxf(tM, p.z);
    tS = tS * __expf(tM - nM) + p.w * __expf(p.z - nM);
    tM = nM;
  }
  lse[t] = make_float2(sM + __logf(sS), tM + __logf(tS));
}

__global__ __launch_bounds__(256)
void finalize(const float* __restrict__ jsdPart, const int* __restrict__ label,
              float* __restrict__ out) {
  __shared__ float fsum[256];
  __shared__ int   cnt[256];
  float s = 0.f;
  for (int i = threadIdx.x; i < kVB * kMB; i += 256) s += jsdPart[i];
  int c = 0;
  for (int i = threadIdx.x; i < kBT; i += 256) c += (label[i] != -100) ? 1 : 0;
  fsum[threadIdx.x] = s;
  cnt[threadIdx.x] = c;
  __syncthreads();
  for (int st = 128; st > 0; st >>= 1) {
    if (threadIdx.x < st) {
      fsum[threadIdx.x] += fsum[threadIdx.x + st];
      cnt[threadIdx.x] += cnt[threadIdx.x + st];
    }
    __syncthreads();
  }
  if (threadIdx.x == 0) {
    const int n = cnt[0] > 0 ? cnt[0] : 1;
    out[0] = fsum[0] / (float)n;
  }
}

extern "C" void kernel_launch(void* const* d_in, const int* in_sizes, int n_in,
                              void* d_out, int out_size, void* d_ws,
                              size_t ws_size, hipStream_t stream) {
  (void)in_sizes; (void)n_in; (void)out_size; (void)ws_size;
  const float* Xs = (const float*)d_in[0];
  const float* Xt = (const float*)d_in[1];
  const float* Ws = (const float*)d_in[2];
  const float* Wt = (const float*)d_in[3];
  const int*   lb = (const int*)d_in[4];
  float* out = (float*)d_out;

  // Workspace layout
  char* base = (char*)d_ws;
  float4* partials = (float4*)base;                                 // 16.38 MB
  float2* lse = (float2*)(base + (size_t)kVB * kBT * sizeof(float4));
  float*  jsdPart = (float*)(base + (size_t)kVB * kBT * sizeof(float4) +
                             (size_t)kBT * sizeof(float2));          // 64 KB

  const dim3 grid(kMB, kVB);  // token blocks fastest -> W stripe reuse in L2
  const dim3 block(256);

  gemm_jsd<1><<<grid, block, 0, stream>>>(Xs, Xt, Ws, Wt, lb, partials, lse,
                                          jsdPart);
  lse_reduce<<<dim3(kBT / 256), block, 0, stream>>>(partials, lse);
  gemm_jsd<2><<<grid, block, 0, stream>>>(Xs, Xt, Ws, Wt, lb, partials, lse,
                                          jsdPart);
  finalize<<<dim3(1), block, 0, stream>>>(jsdPart, lb, out);
}
